// QBWBottleneck_28381143892469
// MI455X (gfx1250) — matrix operand order repaired and checked by reading, not compiled, not run on hardware
//
#include <hip/hip_runtime.h>
#include <hip/hip_bf16.h>

typedef __attribute__((ext_vector_type(16))) _Float16 v16h;
typedef __attribute__((ext_vector_type(8)))  _Float16 v8h;
typedef __attribute__((ext_vector_type(8)))  float    v8f;

#define EPSBN 1e-5f

// ---------------------------------------------------------------------------
// Fragment loaders (wave32 WMMA; this code's contiguous k order, same for A and B)
// A (16x32 f16): lane l -> row m = l&15 ; halves[h]=K (l>>4)*16+h
//                -> two 16B contiguous loads.
// B (32x16 f16): lane l -> col n = l&15 ; halves[h]=K (l>>4)*16+h
//                -> one 32B contiguous load from weights stored [N][K].
// ---------------------------------------------------------------------------
__device__ __forceinline__ v16h load_a_frag(const _Float16* __restrict__ row,
                                            int k0, int lg) {
  union { v16h v; v8h h[2]; } u;
  u.h[0] = *(const v8h*)(row + k0 + lg * 16);
  u.h[1] = *(const v8h*)(row + k0 + lg * 16 + 8);
  return u.v;
}

__device__ __forceinline__ float bn_fq(float acc, float inv, float bias, float sc) {
  float y = acc * inv + bias;
  float q = fminf(fmaxf(rintf(y / sc), -127.f), 127.f) * sc;
  return q;   // forward value of fake_quant is exactly q
}

// ---------------------------------------------------------------------------
// Prep: x (N,C,H,W) f32 -> Xh (N,H*W,C) f16 via LDS tile transpose
// grid (3136/32, 256/32, 32), block 256
// ---------------------------------------------------------------------------
__global__ __launch_bounds__(256) void k_x_to_f16(const float* __restrict__ x,
                                                  _Float16* __restrict__ Xh) {
  __shared__ float tile[32][33];
  const int tx = threadIdx.x & 31, ty = threadIdx.x >> 5;      // ty 0..7
  const int hw0 = blockIdx.x * 32, c0 = blockIdx.y * 32, n = blockIdx.z;
  const float* src = x + ((size_t)n * 256 + c0) * 3136 + hw0;
#pragma unroll
  for (int j = 0; j < 4; ++j) {
    int c = ty * 4 + j;
    tile[c][tx] = src[(size_t)c * 3136 + tx];                  // coalesced read
  }
  __syncthreads();
  _Float16* dst = Xh + ((size_t)n * 3136 + hw0) * 256 + c0;
#pragma unroll
  for (int j = 0; j < 4; ++j) {
    int hw = ty * 4 + j;
    dst[(size_t)hw * 256 + tx] = (_Float16)tile[tx][hw];       // 64B rows
  }
}

// plain f32 -> f16 convert for 1x1 weights (already [O][I] contiguous)
__global__ void k_w_cvt(const float* __restrict__ w, _Float16* __restrict__ o, int n) {
  int i = blockIdx.x * 256 + threadIdx.x;
  if (i < n) o[i] = (_Float16)w[i];
}

// w2 (O=128,I=128,3,3) -> W2h[o][ (ky*3+kx)*128 + i ]  (K-major taps)
__global__ void k_w2_cvt(const float* __restrict__ w, _Float16* __restrict__ o) {
  int t = blockIdx.x * 256 + threadIdx.x;
  if (t >= 128 * 128 * 9) return;
  int kx = t % 3;  int r = t / 3;
  int ky = r % 3;  r /= 3;
  int ic = r % 128;
  int oc = r / 128;
  o[(size_t)oc * 1152 + (size_t)(ky * 3 + kx) * 128 + ic] = (_Float16)w[t];
}

// ---------------------------------------------------------------------------
// conv1: 1x1, M=100352 K=256 N=128, epilogue BN+FQ+ReLU -> O1 f16 [M][128]
// grid (M/64, N/64), block 128 (4 waves); wave = one 64x16 tile (4 WMMA/Kstep)
// ---------------------------------------------------------------------------
__global__ __launch_bounds__(128) void k_conv1(
    const _Float16* __restrict__ Xh, const _Float16* __restrict__ W1,
    const float* __restrict__ g, const float* __restrict__ be,
    const float* __restrict__ mu, const float* __restrict__ va,
    const float* __restrict__ sp, _Float16* __restrict__ O1) {
  const int lane = threadIdx.x & 31, wave = threadIdx.x >> 5;
  const int ln = lane & 15, lg = lane >> 4;
  const int m0 = blockIdx.x * 64;
  const int n0 = (blockIdx.y * 4 + wave) * 16;

  const _Float16* arow[4];
#pragma unroll
  for (int s = 0; s < 4; ++s) arow[s] = Xh + (size_t)(m0 + s * 16 + ln) * 256;
  const _Float16* brow = W1 + (size_t)(n0 + ln) * 256;

  v8f acc[4] = {{}, {}, {}, {}};
  for (int k0 = 0; k0 < 256; k0 += 32) {
    v16h bf = *(const v16h*)(brow + k0 + lg * 16);
#pragma unroll
    for (int s = 0; s < 4; ++s) {
      v16h af = load_a_frag(arow[s], k0, lg);
      acc[s] = __builtin_amdgcn_wmma_f32_16x16x32_f16(false, af, false, bf,
                                                      (short)0, acc[s], false, false);
    }
  }
  const int c = n0 + ln;
  const float inv  = g[c] * rsqrtf(va[c] + EPSBN);
  const float bias = be[c] - mu[c] * inv;
  const float sc   = sp[0];
#pragma unroll
  for (int s = 0; s < 4; ++s)
#pragma unroll
    for (int i = 0; i < 8; ++i) {
      float q = fmaxf(bn_fq(acc[s][i], inv, bias, sc), 0.f);
      O1[(size_t)(m0 + s * 16 + lg * 8 + i) * 128 + c] = (_Float16)q;
    }
}

// ---------------------------------------------------------------------------
// conv2: 3x3 stride2 pad1 implicit GEMM. M=25088 (n*784+oy*28+ox), K=9*128, N=128
// O1 is NHWC f16 on the 56x56 grid. grid (392, 2), block 128.
// ---------------------------------------------------------------------------
__global__ __launch_bounds__(128) void k_conv2(
    const _Float16* __restrict__ O1, const _Float16* __restrict__ W2,
    const float* __restrict__ g, const float* __restrict__ be,
    const float* __restrict__ mu, const float* __restrict__ va,
    const float* __restrict__ sp, _Float16* __restrict__ O2) {
  const int lane = threadIdx.x & 31, wave = threadIdx.x >> 5;
  const int ln = lane & 15, lg = lane >> 4;
  const int m0 = blockIdx.x * 64;
  const int n0 = (blockIdx.y * 4 + wave) * 16;

  int nim[4], oy[4], ox[4];
#pragma unroll
  for (int s = 0; s < 4; ++s) {
    int r = m0 + s * 16 + ln;
    nim[s] = r / 784;
    int t = r - nim[s] * 784;
    oy[s] = t / 28;
    ox[s] = t - oy[s] * 28;
  }
  const _Float16* brow = W2 + (size_t)(n0 + ln) * 1152;

  v8f acc[4] = {{}, {}, {}, {}};
  for (int st = 0; st < 36; ++st) {           // 36 K-steps of 32; tap = st>>2
    const int k0  = st * 32;
    const int tap = st >> 2;
    const int ky  = tap / 3, kx = tap - ky * 3;
    const int kc  = (st & 3) * 32;            // channel offset within this tap
    v16h bf = *(const v16h*)(brow + k0 + lg * 16);
#pragma unroll
    for (int s = 0; s < 4; ++s) {
      int iy = oy[s] * 2 - 1 + ky;
      int ix = ox[s] * 2 - 1 + kx;
      v16h af = {};
      if ((unsigned)iy < 56u && (unsigned)ix < 56u) {
        const _Float16* p =
            O1 + (size_t)(nim[s] * 3136 + iy * 56 + ix) * 128 + kc;
        af = load_a_frag(p, 0, lg);
      }
      acc[s] = __builtin_amdgcn_wmma_f32_16x16x32_f16(false, af, false, bf,
                                                      (short)0, acc[s], false, false);
    }
  }
  const int c = n0 + ln;
  const float inv  = g[c] * rsqrtf(va[c] + EPSBN);
  const float bias = be[c] - mu[c] * inv;
  const float sc   = sp[0];
#pragma unroll
  for (int s = 0; s < 4; ++s)
#pragma unroll
    for (int i = 0; i < 8; ++i) {
      float q = fmaxf(bn_fq(acc[s][i], inv, bias, sc), 0.f);
      O2[(size_t)(m0 + s * 16 + lg * 8 + i) * 128 + c] = (_Float16)q;
    }
}

// ---------------------------------------------------------------------------
// shortcut: 1x1 stride2, M=25088, K=256, N=512. Writes FQ'd BN (no relu)
// as f32 into d_out (NCHW) — consumed by k_conv3. grid (392, 8), block 128.
// ---------------------------------------------------------------------------
__global__ __launch_bounds__(128) void k_convsc(
    const _Float16* __restrict__ Xh, const _Float16* __restrict__ Wsc,
    const float* __restrict__ g, const float* __restrict__ be,
    const float* __restrict__ mu, const float* __restrict__ va,
    const float* __restrict__ sp, float* __restrict__ Out) {
  const int lane = threadIdx.x & 31, wave = threadIdx.x >> 5;
  const int ln = lane & 15, lg = lane >> 4;
  const int m0 = blockIdx.x * 64;
  const int n0 = (blockIdx.y * 4 + wave) * 16;

  const _Float16* arow[4];
#pragma unroll
  for (int s = 0; s < 4; ++s) {
    int r = m0 + s * 16 + ln;
    int ni = r / 784;
    int t = r - ni * 784;
    int yy = t / 28, xx = t - (t / 28) * 28;
    arow[s] = Xh + (size_t)(ni * 3136 + (yy * 2) * 56 + xx * 2) * 256;
  }
  const _Float16* brow = Wsc + (size_t)(n0 + ln) * 256;

  v8f acc[4] = {{}, {}, {}, {}};
  for (int k0 = 0; k0 < 256; k0 += 32) {
    v16h bf = *(const v16h*)(brow + k0 + lg * 16);
#pragma unroll
    for (int s = 0; s < 4; ++s) {
      v16h af = load_a_frag(arow[s], k0, lg);
      acc[s] = __builtin_amdgcn_wmma_f32_16x16x32_f16(false, af, false, bf,
                                                      (short)0, acc[s], false, false);
    }
  }
  const int c = n0 + ln;
  const float inv  = g[c] * rsqrtf(va[c] + EPSBN);
  const float bias = be[c] - mu[c] * inv;
  const float sc   = sp[0];
#pragma unroll
  for (int s = 0; s < 4; ++s)
#pragma unroll
    for (int i = 0; i < 8; ++i) {
      int r  = m0 + s * 16 + lg * 8 + i;
      int ni = r / 784;
      int t  = r - ni * 784;
      int yy = t / 28, xx = t - yy * 28;
      float q = bn_fq(acc[s][i], inv, bias, sc);        // no relu
      Out[(((size_t)ni * 512 + c) * 28 + yy) * 28 + xx] = q;
    }
}

// ---------------------------------------------------------------------------
// conv3: 1x1, M=25088, K=128, N=512. Epilogue: BN+FQ, + shortcut (in d_out),
// ReLU, write fp32 NCHW. grid (392, 8), block 128.
// ---------------------------------------------------------------------------
__global__ __launch_bounds__(128) void k_conv3(
    const _Float16* __restrict__ O2, const _Float16* __restrict__ W3,
    const float* __restrict__ g, const float* __restrict__ be,
    const float* __restrict__ mu, const float* __restrict__ va,
    const float* __restrict__ sp, float* __restrict__ Out) {
  const int lane = threadIdx.x & 31, wave = threadIdx.x >> 5;
  const int ln = lane & 15, lg = lane >> 4;
  const int m0 = blockIdx.x * 64;
  const int n0 = (blockIdx.y * 4 + wave) * 16;

  const _Float16* arow[4];
#pragma unroll
  for (int s = 0; s < 4; ++s) arow[s] = O2 + (size_t)(m0 + s * 16 + ln) * 128;
  const _Float16* brow = W3 + (size_t)(n0 + ln) * 128;

  v8f acc[4] = {{}, {}, {}, {}};
  for (int k0 = 0; k0 < 128; k0 += 32) {
    v16h bf = *(const v16h*)(brow + k0 + lg * 16);
#pragma unroll
    for (int s = 0; s < 4; ++s) {
      v16h af = load_a_frag(arow[s], k0, lg);
      acc[s] = __builtin_amdgcn_wmma_f32_16x16x32_f16(false, af, false, bf,
                                                      (short)0, acc[s], false, false);
    }
  }
  const int c = n0 + ln;
  const float inv  = g[c] * rsqrtf(va[c] + EPSBN);
  const float bias = be[c] - mu[c] * inv;
  const float sc   = sp[0];
#pragma unroll
  for (int s = 0; s < 4; ++s)
#pragma unroll
    for (int i = 0; i < 8; ++i) {
      int r  = m0 + s * 16 + lg * 8 + i;
      int ni = r / 784;
      int t  = r - ni * 784;
      int yy = t / 28, xx = t - yy * 28;
      size_t idx = (((size_t)ni * 512 + c) * 28 + yy) * 28 + xx;
      float q = bn_fq(acc[s][i], inv, bias, sc);
      Out[idx] = fmaxf(q + Out[idx], 0.f);
    }
}

// ---------------------------------------------------------------------------
extern "C" void kernel_launch(void* const* d_in, const int* in_sizes, int n_in,
                              void* d_out, int out_size, void* d_ws, size_t ws_size,
                              hipStream_t stream) {
  const float* x  = (const float*)d_in[0];
  const float* w1 = (const float*)d_in[1];
  const float* g1 = (const float*)d_in[2];
  const float* b1 = (const float*)d_in[3];
  const float* m1 = (const float*)d_in[4];
  const float* v1 = (const float*)d_in[5];
  const float* s1 = (const float*)d_in[6];
  const float* w2 = (const float*)d_in[7];
  const float* g2 = (const float*)d_in[8];
  const float* b2 = (const float*)d_in[9];
  const float* m2 = (const float*)d_in[10];
  const float* v2 = (const float*)d_in[11];
  const float* s2 = (const float*)d_in[12];
  const float* w3 = (const float*)d_in[13];
  const float* g3 = (const float*)d_in[14];
  const float* b3 = (const float*)d_in[15];
  const float* m3 = (const float*)d_in[16];
  const float* v3 = (const float*)d_in[17];
  const float* s3 = (const float*)d_in[18];
  const float* wsw = (const float*)d_in[19];
  const float* gs = (const float*)d_in[20];
  const float* bs = (const float*)d_in[21];
  const float* ms = (const float*)d_in[22];
  const float* vs = (const float*)d_in[23];
  const float* ss = (const float*)d_in[24];
  float* out = (float*)d_out;

  // workspace layout (bytes, all offsets 256B aligned): ~84.3 MB total
  char* ws = (char*)d_ws;
  _Float16* Xh  = (_Float16*)(ws);                  // 100352*256*2 = 51,380,224
  _Float16* O1  = (_Float16*)(ws + 51380224);       // 100352*128*2 = 25,690,112
  _Float16* O2  = (_Float16*)(ws + 77070336);       //  25088*128*2 =  6,422,528
  _Float16* W1h = (_Float16*)(ws + 83492864);       //    128*256*2 =     65,536
  _Float16* W2h = (_Float16*)(ws + 83558400);       //   128*1152*2 =    294,912
  _Float16* W3h = (_Float16*)(ws + 83853312);       //    512*128*2 =    131,072
  _Float16* WSh = (_Float16*)(ws + 83984384);       //    512*256*2 =    262,144

  // --- prep: f32 -> f16, NCHW -> NHWC / [O][K] weight layouts
  k_x_to_f16<<<dim3(98, 8, 32), 256, 0, stream>>>(x, Xh);
  k_w_cvt<<<(128 * 256 + 255) / 256, 256, 0, stream>>>(w1, W1h, 128 * 256);
  k_w2_cvt<<<(128 * 128 * 9 + 255) / 256, 256, 0, stream>>>(w2, W2h);
  k_w_cvt<<<(512 * 128 + 255) / 256, 256, 0, stream>>>(w3, W3h, 512 * 128);
  k_w_cvt<<<(512 * 256 + 255) / 256, 256, 0, stream>>>(wsw, WSh, 512 * 256);

  // --- fused WMMA implicit-GEMM pipeline
  k_conv1 <<<dim3(1568, 2), 128, 0, stream>>>(Xh, W1h, g1, b1, m1, v1, s1, O1);
  k_conv2 <<<dim3(392, 2),  128, 0, stream>>>(O1, W2h, g2, b2, m2, v2, s2, O2);
  k_convsc<<<dim3(392, 8),  128, 0, stream>>>(Xh, WSh, gs, bs, ms, vs, ss, out);
  k_conv3 <<<dim3(392, 8),  128, 0, stream>>>(O2, W3h, g3, b3, m3, v3, s3, out);
}
